// ConvSelfAttention_43576738185969
// MI455X (gfx1250) — compile-verified, matrix operand order checked by reading, not run on hardware
//
#include <hip/hip_runtime.h>

// ---------- types ----------
typedef __attribute__((ext_vector_type(16))) __bf16 v16bf;
typedef __attribute__((ext_vector_type(8)))  __bf16 v8bf;
typedef __attribute__((ext_vector_type(2)))  __bf16 v2bf;
typedef __attribute__((ext_vector_type(8)))  float  v8f;
typedef __attribute__((ext_vector_type(4)))  float  v4f;

constexpr int Bb  = 4;
constexpr int Cc  = 512;
constexpr int Nn  = 2048;
constexpr int Hh  = 8;
constexpr int Dd  = 64;
constexpr int HID = 512;
constexpr float SCALE = 0.125f;   // 64^-0.5

#define DEVINL __device__ __forceinline__

// Load a 16-element WMMA fragment (A-layout / B^T-row layout) from a
// contiguous K-run of bf16. half = lane>>4.
// elements 0..7  = p[half*8 .. half*8+7]
// elements 8..15 = p[16+half*8 .. 16+half*8+7]
DEVINL v16bf frag16(const __bf16* p, int half) {
  v8bf lo = *(const v8bf*)(p + half * 8);
  v8bf hi = *(const v8bf*)(p + 16 + half * 8);
  v16bf r;
#pragma unroll
  for (int i = 0; i < 8; ++i) { r[i] = lo[i]; r[8 + i] = hi[i]; }
  return r;
}

DEVINL v8f wmma_bf16(v16bf a, v16bf b, v8f c) {
  return __builtin_amdgcn_wmma_f32_16x16x32_bf16(
      /*neg_a=*/false, a, /*neg_b=*/false, b,
      /*c_mod=*/(short)0, c, /*reuse_a=*/false, /*reuse_b=*/false);
}

// Async copy 16B from global to LDS (CDNA5 async engine, ASYNCcnt-tracked).
DEVINL void async_ld16(unsigned lds_byte_off, const __bf16* g) {
  asm volatile("global_load_async_to_lds_b128 %0, %1, off"
               :: "v"(lds_byte_off), "v"(g) : "memory");
}
DEVINL void wait_async0() {
  asm volatile("s_wait_asynccnt 0x0" ::: "memory");
}
DEVINL unsigned lds_off(const void* p) {
  // generic LDS address: low 32 bits are the LDS byte offset
  return (unsigned)(uintptr_t)p;
}

// =====================================================================
// Kernel 1: qkv = w_qkv(1536x512) @ x[b](512x2048), f32 in -> bf16 out.
// Block tile 128x128, wave tile 32x64 (8 wmmas / k-step), K staged 32.
// Writes qT/kT as [B,H,N,D] (q pre-scaled), v as [B,H,D,N].
// =====================================================================
__global__ __launch_bounds__(256)
void qkv_gemm(const float* __restrict__ x, const float* __restrict__ w,
              __bf16* __restrict__ qT, __bf16* __restrict__ kT,
              __bf16* __restrict__ vv) {
  __shared__ __align__(16) __bf16 As[128 * 32];   // As[r][kk]
  __shared__ __align__(16) __bf16 Bs[128 * 32];   // Bs[col][kk] (B transposed)
  const int tid  = threadIdx.x;
  const int wv   = tid >> 5, lane = tid & 31;
  const int half = lane >> 4, lc = lane & 15;
  const int n0 = blockIdx.x * 128, o0 = blockIdx.y * 128, b = blockIdx.z;
  const float* xb = x + (size_t)b * Cc * Nn;
  const int mr = (wv & 3) * 32, nc = (wv >> 2) * 64;
  // staging indices
  const int ar = tid >> 1,        ak = (tid & 1) * 16;   // A: 16 f32 / thread
  const int bk = (tid >> 4) * 2,  bc = (tid & 15) * 8;   // B: 2 rows x 8 cols

  v8f acc[2][4] = {};
  for (int k0 = 0; k0 < Cc; k0 += 32) {
    { // stage A: 16 contiguous f32 -> 16 bf16
      const float* src = w + (size_t)(o0 + ar) * Cc + k0 + ak;
      v4f f0 = ((const v4f*)src)[0], f1 = ((const v4f*)src)[1];
      v4f f2 = ((const v4f*)src)[2], f3 = ((const v4f*)src)[3];
      v8bf h0, h1;
#pragma unroll
      for (int i = 0; i < 4; ++i) {
        h0[i] = (__bf16)f0[i]; h0[4 + i] = (__bf16)f1[i];
        h1[i] = (__bf16)f2[i]; h1[4 + i] = (__bf16)f3[i];
      }
      *(v8bf*)(As + ar * 32 + ak)     = h0;
      *(v8bf*)(As + ar * 32 + ak + 8) = h1;
    }
    { // stage B transposed: two k-rows packed per b32 store
      const float* s0 = xb + (size_t)(k0 + bk) * Nn + n0 + bc;
      const float* s1 = s0 + Nn;
      v4f r0a = ((const v4f*)s0)[0], r0b = ((const v4f*)s0)[1];
      v4f r1a = ((const v4f*)s1)[0], r1b = ((const v4f*)s1)[1];
#pragma unroll
      for (int i = 0; i < 4; ++i) {
        v2bf p0; p0[0] = (__bf16)r0a[i]; p0[1] = (__bf16)r1a[i];
        *(v2bf*)(Bs + (bc + i) * 32 + bk) = p0;
        v2bf p1; p1[0] = (__bf16)r0b[i]; p1[1] = (__bf16)r1b[i];
        *(v2bf*)(Bs + (bc + 4 + i) * 32 + bk) = p1;
      }
    }
    __syncthreads();
    v16bf a0 = frag16(As + (mr + lc) * 32, half);
    v16bf a1 = frag16(As + (mr + 16 + lc) * 32, half);
#pragma unroll
    for (int t = 0; t < 4; ++t) {
      v16bf bt = frag16(Bs + (nc + t * 16 + lc) * 32, half);
      acc[0][t] = wmma_bf16(a0, bt, acc[0][t]);
      acc[1][t] = wmma_bf16(a1, bt, acc[1][t]);
    }
    __syncthreads();
  }

#pragma unroll
  for (int i = 0; i < 2; ++i) {
#pragma unroll
    for (int t = 0; t < 4; ++t) {
      int n = n0 + nc + t * 16 + lc;
#pragma unroll
      for (int e = 0; e < 8; ++e) {
        int o = o0 + mr + i * 16 + e + 8 * half;
        float val = acc[i][t][e];
        if (o < HID) {                               // Q -> qT[b][h][n][d] * SCALE
          int h = o >> 6, d = o & 63;
          qT[(((size_t)b * Hh + h) * Nn + n) * Dd + d] = (__bf16)(val * SCALE);
        } else if (o < 2 * HID) {                    // K -> kT[b][h][n][d]
          int o2 = o - HID; int h = o2 >> 6, d = o2 & 63;
          kT[(((size_t)b * Hh + h) * Nn + n) * Dd + d] = (__bf16)val;
        } else {                                     // V -> v[b][h][d][n]
          int o2 = o - 2 * HID; int h = o2 >> 6, d = o2 & 63;
          vv[(((size_t)b * Hh + h) * Dd + d) * Nn + n] = (__bf16)val;
        }
      }
    }
  }
}

// =====================================================================
// Kernel 2: flash attention. Block = (b,h) x 128 query rows; each of 8
// waves owns 16 rows. K/V tiles staged with async global->LDS copies
// (ASYNCcnt), online softmax, P routed through LDS (C->A layout),
// O accumulated in f32 WMMA accumulators. Writes ao[b][h*64+d][n] bf16.
// =====================================================================
__global__ __launch_bounds__(256)
void flash_attn(const __bf16* __restrict__ qT, const __bf16* __restrict__ kT,
                const __bf16* __restrict__ vv, __bf16* __restrict__ ao) {
  __shared__ __align__(16) __bf16 Ks[64 * 64];        // Ks[j][d]
  __shared__ __align__(16) __bf16 Vs[64 * 64];        // Vs[d][j]
  __shared__ __align__(16) __bf16 Ps[8 * 16 * 64];    // per-wave P patch
  const int tid  = threadIdx.x;
  const int wv   = tid >> 5, lane = tid & 31;
  const int half = lane >> 4, lc = lane & 15;
  const int m0 = blockIdx.x * 128;
  const int bh = blockIdx.y;                          // b*H + h
  const __bf16* qB = qT + (size_t)bh * Nn * Dd;
  const __bf16* kB = kT + (size_t)bh * Nn * Dd;
  const __bf16* vB = vv + (size_t)bh * Dd * Nn;

  const int iA = m0 + wv * 16 + lc;                   // this lane's A row
  v16bf aq0 = frag16(qB + (size_t)iA * Dd, half);       // d = 0..31
  v16bf aq1 = frag16(qB + (size_t)iA * Dd + 32, half);  // d = 32..63

  // per-tile staging addresses (loop-invariant parts)
  const int kj = tid >> 2, kd = (tid & 3) * 16;       // K: Ks[j][d]
  const int vd = tid >> 2, vj = (tid & 3) * 16;       // V: Vs[d][j]
  const unsigned ksOff = lds_off(Ks + kj * 64 + kd);
  const unsigned vsOff = lds_off(Vs + vd * 64 + vj);

  float mrow[8], lrow[8];
#pragma unroll
  for (int e = 0; e < 8; ++e) { mrow[e] = -1e30f; lrow[e] = 0.f; }
  v8f oacc[4] = {};                                   // d-subtiles 0,16,32,48

  for (int j0 = 0; j0 < Nn; j0 += 64) {
    { // async stage K tile (bf16 copy, no VGPR round trip)
      const __bf16* g = kB + (size_t)(j0 + kj) * Dd + kd;
      async_ld16(ksOff,      g);
      async_ld16(ksOff + 16, g + 8);
    }
    { // async stage V tile
      const __bf16* g = vB + (size_t)vd * Nn + j0 + vj;
      async_ld16(vsOff,      g);
      async_ld16(vsOff + 16, g + 8);
    }
    wait_async0();
    __syncthreads();

    // S = (Q*scale) K^T : 16x64 in four 16x16 accumulators
    v8f s[4] = {};
#pragma unroll
    for (int jt = 0; jt < 4; ++jt) {
      const __bf16* bp = Ks + (jt * 16 + lc) * 64;
      s[jt] = wmma_bf16(aq0, frag16(bp, half), s[jt]);
      s[jt] = wmma_bf16(aq1, frag16(bp + 32, half), s[jt]);
    }

    // online softmax: row e lives across one 16-lane group
#pragma unroll
    for (int e = 0; e < 8; ++e) {
      float tm = fmaxf(fmaxf(s[0][e], s[1][e]), fmaxf(s[2][e], s[3][e]));
#pragma unroll
      for (int off = 8; off >= 1; off >>= 1)
        tm = fmaxf(tm, __shfl_xor(tm, off, 16));
      float mnew = fmaxf(mrow[e], tm);
      float corr = __expf(mrow[e] - mnew);
      float psum = 0.f;
#pragma unroll
      for (int jt = 0; jt < 4; ++jt) {
        float p = __expf(s[jt][e] - mnew);
        s[jt][e] = p;
        psum += p;
      }
#pragma unroll
      for (int off = 8; off >= 1; off >>= 1)
        psum += __shfl_xor(psum, off, 16);
      lrow[e] = lrow[e] * corr + psum;
      mrow[e] = mnew;
#pragma unroll
      for (int dt = 0; dt < 4; ++dt) oacc[dt][e] *= corr;
    }

    // C-layout -> A-layout for P via per-wave LDS patch
    __bf16* pw = Ps + wv * 16 * 64;
#pragma unroll
    for (int jt = 0; jt < 4; ++jt) {
      int j = jt * 16 + lc;
#pragma unroll
      for (int e = 0; e < 8; ++e)
        pw[(e + 8 * half) * 64 + j] = (__bf16)s[jt][e];
    }
    const __bf16* pr = pw + lc * 64;
    v16bf ap0 = frag16(pr, half);
    v16bf ap1 = frag16(pr + 32, half);

    // O += P @ V^T
#pragma unroll
    for (int dt = 0; dt < 4; ++dt) {
      const __bf16* bp = Vs + (dt * 16 + lc) * 64;
      oacc[dt] = wmma_bf16(ap0, frag16(bp, half), oacc[dt]);
      oacc[dt] = wmma_bf16(ap1, frag16(bp + 32, half), oacc[dt]);
    }
    __syncthreads();
  }

  // normalize and store: ao[b][h*64+d][n]
  const int bb = bh >> 3, hh = bh & 7;
  __bf16* aoB = ao + ((size_t)bb * HID + (size_t)hh * 64) * Nn;
#pragma unroll
  for (int e = 0; e < 8; ++e) {
    float inv = 1.f / lrow[e];
    int i = m0 + wv * 16 + e + 8 * half;
#pragma unroll
    for (int dt = 0; dt < 4; ++dt) {
      int d = dt * 16 + lc;
      aoB[(size_t)d * Nn + i] = (__bf16)(oacc[dt][e] * inv);
    }
  }
}

// =====================================================================
// Kernel 3: y[b] = w_out(512x512) @ ao[b](512x2048) + b_out, f32 out.
// Same 128x128 block / 32x64 wave tiling as kernel 1.
// =====================================================================
__global__ __launch_bounds__(256)
void out_gemm(const __bf16* __restrict__ ao, const float* __restrict__ w,
              const float* __restrict__ bias, float* __restrict__ y) {
  __shared__ __align__(16) __bf16 As[128 * 32];
  __shared__ __align__(16) __bf16 Bs[128 * 32];
  const int tid  = threadIdx.x;
  const int wv   = tid >> 5, lane = tid & 31;
  const int half = lane >> 4, lc = lane & 15;
  const int n0 = blockIdx.x * 128, c0 = blockIdx.y * 128, b = blockIdx.z;
  const __bf16* aob = ao + (size_t)b * HID * Nn;
  const int mr = (wv & 3) * 32, nc = (wv >> 2) * 64;
  const int ar = tid >> 1,       ak = (tid & 1) * 16;
  const int bk = (tid >> 4) * 2, bc = (tid & 15) * 8;

  v8f acc[2][4] = {};
  for (int k0 = 0; k0 < HID; k0 += 32) {
    { // stage A (w_out, f32 -> bf16)
      const float* src = w + (size_t)(c0 + ar) * HID + k0 + ak;
      v4f f0 = ((const v4f*)src)[0], f1 = ((const v4f*)src)[1];
      v4f f2 = ((const v4f*)src)[2], f3 = ((const v4f*)src)[3];
      v8bf h0, h1;
#pragma unroll
      for (int i = 0; i < 4; ++i) {
        h0[i] = (__bf16)f0[i]; h0[4 + i] = (__bf16)f1[i];
        h1[i] = (__bf16)f2[i]; h1[4 + i] = (__bf16)f3[i];
      }
      *(v8bf*)(As + ar * 32 + ak)     = h0;
      *(v8bf*)(As + ar * 32 + ak + 8) = h1;
    }
    { // stage B transposed (ao already bf16): pack 2 k-rows per b32
      const __bf16* s0 = aob + (size_t)(k0 + bk) * Nn + n0 + bc;
      v8bf r0 = *(const v8bf*)s0;
      v8bf r1 = *(const v8bf*)(s0 + Nn);
#pragma unroll
      for (int i = 0; i < 8; ++i) {
        v2bf p; p[0] = r0[i]; p[1] = r1[i];
        *(v2bf*)(Bs + (bc + i) * 32 + bk) = p;
      }
    }
    __syncthreads();
    v16bf a0 = frag16(As + (mr + lc) * 32, half);
    v16bf a1 = frag16(As + (mr + 16 + lc) * 32, half);
#pragma unroll
    for (int t = 0; t < 4; ++t) {
      v16bf bt = frag16(Bs + (nc + t * 16 + lc) * 32, half);
      acc[0][t] = wmma_bf16(a0, bt, acc[0][t]);
      acc[1][t] = wmma_bf16(a1, bt, acc[1][t]);
    }
    __syncthreads();
  }

#pragma unroll
  for (int i = 0; i < 2; ++i) {
#pragma unroll
    for (int t = 0; t < 4; ++t) {
      int n = n0 + nc + t * 16 + lc;
#pragma unroll
      for (int e = 0; e < 8; ++e) {
        int c = c0 + mr + i * 16 + e + 8 * half;
        y[((size_t)b * Cc + c) * Nn + n] = acc[i][t][e] + bias[c];
      }
    }
  }
}

// =====================================================================
extern "C" void kernel_launch(void* const* d_in, const int* in_sizes, int n_in,
                              void* d_out, int out_size, void* d_ws, size_t ws_size,
                              hipStream_t stream) {
  const float* x     = (const float*)d_in[0];
  const float* w_qkv = (const float*)d_in[1];
  const float* w_out = (const float*)d_in[2];
  const float* b_out = (const float*)d_in[3];
  float* y = (float*)d_out;

  const size_t per = (size_t)Bb * Hh * Nn * Dd;   // 4M bf16 elements each
  __bf16* qT = (__bf16*)d_ws;
  __bf16* kT = qT + per;
  __bf16* vv = kT + per;
  __bf16* ao = vv + per;                          // [B][HID][N] bf16
  // total ws use: 4 * per * 2 bytes = 32 MiB

  qkv_gemm<<<dim3(Nn / 128, (3 * HID) / 128, Bb), 256, 0, stream>>>(x, w_qkv, qT, kT, vv);
  flash_attn<<<dim3(Nn / 128, Bb * Hh), 256, 0, stream>>>(qT, kT, vv, ao);
  out_gemm<<<dim3(Nn / 128, Cc / 128, Bb), 256, 0, stream>>>(ao, w_out, b_out, y);
}
